// CompositionalGatedRecurrence_34316788695878
// MI455X (gfx1250) — matrix operand order repaired and checked by reading, not compiled, not run on hardware
//
#include <hip/hip_runtime.h>
#include <hip/hip_bf16.h>

// ---------------------------------------------------------------------------
// Problem constants (from reference)
// ---------------------------------------------------------------------------
#define BB 4
#define SS 2048
#define DM 1024
#define NH 16
#define DH 64
#define HID 1024
#define NPRIM 16
#define RANK 256
#define TOPK 4
#define BSTOT (BB * SS)          // 8192 token rows
#define NCAT 5120                // 4*1024 proj cols + 1024 out-gate cols

typedef __attribute__((ext_vector_type(16))) __bf16 v16bf;
typedef __attribute__((ext_vector_type(8)))  float  v8f;

// ---------------------------------------------------------------------------
// Async copy to LDS (CDNA5): guarded, with synchronous fallback
// ---------------------------------------------------------------------------
#if defined(__has_builtin)
#if __has_builtin(__builtin_amdgcn_global_load_async_to_lds_b128)
#define ASYNC_COPY 1
#endif
#endif
#ifndef ASYNC_COPY
#define ASYNC_COPY 0
#endif

#if ASYNC_COPY
// The builtin's parameters are '__vector_size__(16) int' pointers in
// addrspace(1) (global) and addrspace(3) (LDS).
typedef int v4i_vs __attribute__((vector_size(16)));
typedef __attribute__((address_space(1))) v4i_vs gv4i;
typedef __attribute__((address_space(3))) v4i_vs lv4i;
__device__ __forceinline__ void async_cp16(const void* g, void* l) {
  __builtin_amdgcn_global_load_async_to_lds_b128(
      (gv4i*)(unsigned long long)(size_t)g,
      (lv4i*)(unsigned int)(size_t)l, 0, 0);
}
// wait until at most N of this wave's async copies remain outstanding
__device__ __forceinline__ void wait_async_le0() {
#if __has_builtin(__builtin_amdgcn_s_wait_asynccnt)
  __builtin_amdgcn_s_wait_asynccnt(0);
#else
  asm volatile("s_wait_asynccnt 0" ::: "memory");
#endif
}
__device__ __forceinline__ void wait_async_le4() {
#if __has_builtin(__builtin_amdgcn_s_wait_asynccnt)
  __builtin_amdgcn_s_wait_asynccnt(4);
#else
  asm volatile("s_wait_asynccnt 4" ::: "memory");
#endif
}
#endif

// ---------------------------------------------------------------------------
// Generic bf16 WMMA GEMM: C[M,N] = A[M,K] * Bt[N,K]^T
//   A  : bf16 row-major [M][lda]
//   Bt : bf16 row-major [N][ldb]   (i.e. B stored transposed -> contiguous K)
//   out: fp32 C[..][ldc] or bf16 Cb[..][ldc]
// 64x64 tile per block of 128 threads (4 waves); KSTEP = 32.
// Double-buffered LDS: async copy of tile k+1 overlaps WMMAs on tile k.
// ---------------------------------------------------------------------------
#define TILE 64
#define KSTEP 32
#define LDSS 48   // LDS row stride in elems: 96 B -> keeps 16 B alignment

__global__ __launch_bounds__(128)
void gemm_bf16_wmma(const __bf16* __restrict__ A, int lda,
                    const __bf16* __restrict__ Bt, int ldb,
                    float* __restrict__ C, __bf16* __restrict__ Cb,
                    int ldc, int K)
{
  __shared__ __align__(16) __bf16 sA[2][TILE][LDSS];
  __shared__ __align__(16) __bf16 sB[2][TILE][LDSS];

  const int tid  = threadIdx.x;
  const int wave = tid >> 5;
  const int lane = tid & 31;
  const int m0 = blockIdx.y * TILE;
  const int n0 = blockIdx.x * TILE;

  // stage one 64x32 A-tile + 64x32 B-tile into LDS buffer `buf`
  // (4 async b128 copies per thread per call)
  auto stage = [&](int buf, int k0) {
#pragma unroll
    for (int i = 0; i < 2; ++i) {
      int linear = tid + i * 128;              // 0..255
      int row = linear >> 2;                   // 0..63
      int seg = (linear & 3) * 8;              // elem offset: 0,8,16,24
      const __bf16* ga = A  + (size_t)(m0 + row) * lda + k0 + seg;
      const __bf16* gb = Bt + (size_t)(n0 + row) * ldb + k0 + seg;
#if ASYNC_COPY
      async_cp16(ga, &sA[buf][row][seg]);
      async_cp16(gb, &sB[buf][row][seg]);
#else
      *(uint4*)&sA[buf][row][seg] = *(const uint4*)ga;
      *(uint4*)&sB[buf][row][seg] = *(const uint4*)gb;
#endif
    }
  };

  v8f acc[4] = {};
  const int nk = K / KSTEP;

  stage(0, 0);
  for (int kt = 0; kt < nk; ++kt) {
    const int cur = kt & 1;
    if (kt + 1 < nk) {
      stage(cur ^ 1, (kt + 1) * KSTEP);        // prefetch next tile (async)
#if ASYNC_COPY
      wait_async_le4();                        // oldest 4 (= current buf) done
#endif
    } else {
#if ASYNC_COPY
      wait_async_le0();
#endif
    }
    __syncthreads();                           // current buffer visible to all

    // ---- A fragment: wave owns rows [wave*16, wave*16+16) ----
    // k order used here (B matches): lane<16 -> K {0..15}; lane>=16 -> K {16..31}
    const int mrow = (wave << 4) + (lane & 15);
    const int koff = (lane >> 4) << 4;         // 0 or 16
    v16bf afrag;
    {
      const __bf16* pa = &sA[cur][mrow][0];
#pragma unroll
      for (int i = 0; i < 8; ++i) {
        afrag[i]     = pa[koff + i];
        afrag[8 + i] = pa[8 + koff + i];
      }
    }

    // ---- load ALL B fragments first, then issue WMMAs back-to-back ----
    const int kbase = (lane >> 4) << 4;        // 0 or 16
    const int ncol  = lane & 15;
    v16bf bfr[4];
#pragma unroll
    for (int j = 0; j < 4; ++j) {
      const __bf16* pb = &sB[cur][(j << 4) + ncol][kbase];
#pragma unroll
      for (int i = 0; i < 16; ++i) bfr[j][i] = pb[i];
    }
#pragma unroll
    for (int j = 0; j < 4; ++j) {
      acc[j] = __builtin_amdgcn_wmma_f32_16x16x32_bf16(
          false, afrag, false, bfr[j], (short)0, acc[j], false, false);
    }
    __syncthreads();                           // reads done before buf reuse
  }

  // ---- store: C layout VGPR r -> M = r + (lane>=16)*8, N = lane&15 ----
  const int nlo = lane & 15;
  const int mhi = (lane >> 4) << 3;
#pragma unroll
  for (int j = 0; j < 4; ++j) {
#pragma unroll
    for (int r = 0; r < 8; ++r) {
      size_t off = (size_t)(m0 + (wave << 4) + mhi + r) * ldc
                 + n0 + (j << 4) + nlo;
      if (Cb) Cb[off] = (__bf16)acc[j][r];
      else    C[off]  = acc[j][r];
    }
  }
}

// ---------------------------------------------------------------------------
// Top-k softmax weights for the 4 banks (q,k,v,gate)
// ---------------------------------------------------------------------------
__global__ void topk_kernel(const float* __restrict__ ql,
                            const float* __restrict__ kl,
                            const float* __restrict__ vl,
                            const float* __restrict__ gl,
                            int* __restrict__ idx_out,
                            float* __restrict__ w_out)
{
  int bank = threadIdx.x;
  if (bank >= 4) return;
  const float* l = bank == 0 ? ql : bank == 1 ? kl : bank == 2 ? vl : gl;
  float v[NPRIM];
  float mx = -1e30f;
  for (int i = 0; i < NPRIM; ++i) { v[i] = l[i]; mx = fmaxf(mx, v[i]); }
  float sum = 0.f;
  for (int i = 0; i < NPRIM; ++i) { v[i] = expf(v[i] - mx); sum += v[i]; }
  for (int i = 0; i < NPRIM; ++i) v[i] /= sum;
  int   ids[TOPK];
  float vals[TOPK];
  bool  used[NPRIM] = {};
  for (int t = 0; t < TOPK; ++t) {
    int   bi = -1;
    float bv = -1.f;
    for (int i = 0; i < NPRIM; ++i)
      if (!used[i] && v[i] > bv) { bv = v[i]; bi = i; }
    used[bi] = true; ids[t] = bi; vals[t] = bv;
  }
  float s = vals[0] + vals[1] + vals[2] + vals[3];
  for (int t = 0; t < TOPK; ++t) {
    idx_out[bank * TOPK + t] = ids[t];
    w_out[bank * TOPK + t]   = vals[t] / s;
  }
}

// ---------------------------------------------------------------------------
// Gather/convert kernels for the collapsed bank matrices
//   UcatBt[bank][d][j*256+r] = bf16( U_bank[idx][d][r] )      (GEMM B^T)
//   VcatA [bank][o][j*256+r] = bf16( w_j * V_bank[idx][r][o] ) (GEMM A)
// ---------------------------------------------------------------------------
__global__ void gather_u_kernel(const float* __restrict__ qU,
                                const float* __restrict__ kU,
                                const float* __restrict__ vU,
                                const int* __restrict__ idx,
                                __bf16* __restrict__ out)
{
  size_t t = (size_t)blockIdx.x * blockDim.x + threadIdx.x;
  if (t >= (size_t)4 * DM * (TOPK * RANK)) return;
  int bank = (int)(t >> 20);
  int rem  = (int)(t & 0xFFFFF);
  int d    = rem >> 10;
  int jr   = rem & 1023;
  int j = jr >> 8, r = jr & 255;
  const float* U = bank == 0 ? qU : bank == 1 ? kU : vU;  // banks 2,3 use v_U
  int p = idx[bank * TOPK + j];
  out[t] = (__bf16)U[((size_t)p * DM + d) * RANK + r];
}

__global__ void gather_v_kernel(const float* __restrict__ qV,
                                const float* __restrict__ kV,
                                const float* __restrict__ vV,
                                const int* __restrict__ idx,
                                const float* __restrict__ w,
                                __bf16* __restrict__ out)
{
  size_t t = (size_t)blockIdx.x * blockDim.x + threadIdx.x;
  if (t >= (size_t)4 * HID * (TOPK * RANK)) return;
  int bank = (int)(t >> 20);
  int rem  = (int)(t & 0xFFFFF);
  int o    = rem >> 10;
  int jr   = rem & 1023;
  int j = jr >> 8, r = jr & 255;
  const float* V = bank == 0 ? qV : bank == 1 ? kV : vV;  // banks 2,3 use v_V
  int p = idx[bank * TOPK + j];
  out[t] = (__bf16)(w[bank * TOPK + j] * V[((size_t)p * RANK + r) * HID + o]);
}

// fp32 -> bf16 flat convert
__global__ void cvt_bf16_kernel(const float* __restrict__ in,
                                __bf16* __restrict__ out, size_t n)
{
  size_t t = (size_t)blockIdx.x * blockDim.x + threadIdx.x;
  if (t < n) out[t] = (__bf16)in[t];
}

// ---------------------------------------------------------------------------
// Decay: ld[bs][h] = -softplus( dot(x[bs], decay_w[h]) + decay_b[h] )
// ---------------------------------------------------------------------------
__global__ __launch_bounds__(256)
void decay_kernel(const float* __restrict__ x,
                  const float* __restrict__ dw,
                  const float* __restrict__ db,
                  float* __restrict__ ld)
{
  __shared__ float sx[DM];
  int bs = blockIdx.x;
  for (int i = threadIdx.x; i < DM; i += 256) sx[i] = x[(size_t)bs * DM + i];
  __syncthreads();
  int h = threadIdx.x >> 4;       // 0..15
  int part = threadIdx.x & 15;    // 0..15
  float acc = 0.f;
  for (int i = part; i < DM; i += 16) acc += sx[i] * dw[h * DM + i];
#pragma unroll
  for (int off = 8; off > 0; off >>= 1) acc += __shfl_xor(acc, off, 16);
  if (part == 0) {
    float z  = acc + db[h];
    float sp = (z > 20.f) ? z : log1pf(expf(z));
    ld[(size_t)bs * NH + h] = -sp;
  }
}

// ---------------------------------------------------------------------------
// Pointwise: kv = sigmoid(gate) * k * v  (written in place into v's slot)
// Y row layout (ldc = NCAT): [q | k | v | gate | ogate]
// ---------------------------------------------------------------------------
__global__ void kv_kernel(float* __restrict__ Y)
{
  size_t t = (size_t)blockIdx.x * blockDim.x + threadIdx.x;
  if (t >= (size_t)BSTOT * HID) return;
  size_t bs = t >> 10;
  int o = (int)(t & 1023);
  float* row = Y + bs * NCAT;
  float k = row[HID + o];
  float v = row[2 * HID + o];
  float g = row[3 * HID + o];
  row[2 * HID + o] = k * v * (1.f / (1.f + expf(-g)));
}

// ---------------------------------------------------------------------------
// Decay-gated recurrence: state = exp(ld)*state + kv ; out = q*state
// (mathematically identical to the reference's chunked scan)
// one thread per (b,h,d) lane, sequential over S
// ---------------------------------------------------------------------------
__global__ __launch_bounds__(64)
void scan_kernel(const float* __restrict__ Y,
                 const float* __restrict__ LD,
                 float* __restrict__ out)
{
  int bh = blockIdx.x;            // b*NH + h
  int b = bh >> 4, h = bh & 15;
  int d = threadIdx.x;            // 0..63
  int o = h * DH + d;
  float state = 0.f;
  for (int s = 0; s < SS; ++s) {
    size_t bs = (size_t)b * SS + s;
    float a = __expf(LD[bs * NH + h]);
    const float* row = Y + bs * NCAT;
    state = a * state + row[2 * HID + o];
    out[bs * HID + o] = row[o] * state;
  }
}

// ---------------------------------------------------------------------------
// RMS norm over HID + out-gate sigmoid, write bf16 for final WMMA GEMM
// ---------------------------------------------------------------------------
__global__ __launch_bounds__(256)
void rms_gate_kernel(const float* __restrict__ S,
                     const float* __restrict__ Y,
                     const float* __restrict__ rms_w,
                     __bf16* __restrict__ Ob)
{
  __shared__ float red[8];
  int bs = blockIdx.x;
  const float* s = S + (size_t)bs * HID;
  float acc = 0.f;
  for (int i = threadIdx.x; i < HID; i += 256) { float v = s[i]; acc += v * v; }
#pragma unroll
  for (int off = 16; off > 0; off >>= 1) acc += __shfl_xor(acc, off, 32);
  int wave = threadIdx.x >> 5, lane = threadIdx.x & 31;
  if (lane == 0) red[wave] = acc;
  __syncthreads();
  if (threadIdx.x == 0) {
    float t = 0.f;
    for (int w = 0; w < 8; ++w) t += red[w];
    red[0] = rsqrtf(t / (float)HID + 1.1920929e-7f);
  }
  __syncthreads();
  float scale = red[0];
  const float* og = Y + (size_t)bs * NCAT + 4 * HID;
  for (int i = threadIdx.x; i < HID; i += 256) {
    float v = s[i] * scale * rms_w[i];
    v *= 1.f / (1.f + expf(-og[i]));
    Ob[(size_t)bs * HID + i] = (__bf16)v;
  }
}

// ---------------------------------------------------------------------------
// Host-side orchestration
// ---------------------------------------------------------------------------
extern "C" void kernel_launch(void* const* d_in, const int* in_sizes, int n_in,
                              void* d_out, int out_size, void* d_ws, size_t ws_size,
                              hipStream_t stream)
{
  (void)in_sizes; (void)n_in; (void)out_size; (void)ws_size;

  const float* x      = (const float*)d_in[0];
  const float* q_U    = (const float*)d_in[1];
  const float* q_V    = (const float*)d_in[2];
  const float* k_U    = (const float*)d_in[3];
  const float* k_V    = (const float*)d_in[4];
  const float* v_U    = (const float*)d_in[5];
  const float* v_V    = (const float*)d_in[6];
  const float* q_lg   = (const float*)d_in[7];
  const float* k_lg   = (const float*)d_in[8];
  const float* v_lg   = (const float*)d_in[9];
  const float* g_lg   = (const float*)d_in[10];
  const float* dec_w  = (const float*)d_in[11];
  const float* dec_b  = (const float*)d_in[12];
  const float* og_w   = (const float*)d_in[13];
  const float* op_w   = (const float*)d_in[14];
  const float* rms_w  = (const float*)d_in[15];

  char* ws = (char*)d_ws;
  // workspace layout (bytes)
  const size_t off_hdr  = 0;                                   // idx + weights
  const size_t off_Xb   = 1024;                                // 16 MB bf16
  const size_t off_Wcat = off_Xb   + (size_t)BSTOT * DM * 2;   // 10 MB bf16 [NCAT][DM]
  const size_t off_PBt  = off_Wcat + (size_t)NCAT * DM * 2;    //  2 MB bf16 [DM][HID]
  const size_t off_LD   = off_PBt  + (size_t)DM * HID * 2;     //  0.5 MB f32
  const size_t off_S    = off_LD   + (size_t)BSTOT * NH * 4;   // 32 MB f32 scan out
  const size_t off_Ob   = off_S    + (size_t)BSTOT * HID * 4;  // 16 MB bf16
  const size_t off_Y    = off_Ob   + (size_t)BSTOT * HID * 2;  // 160 MB f32 [BSTOT][NCAT]
  // Ucat/Vcat alias the Y region (consumed before Y is written)
  const size_t off_Ucat = off_Y;
  const size_t off_Vcat = off_Y + (size_t)4 * DM * 1024 * 2;

  int*    idx  = (int*)(ws + off_hdr);
  float*  wts  = (float*)(ws + off_hdr + 256);
  __bf16* Xb   = (__bf16*)(ws + off_Xb);
  __bf16* Wcat = (__bf16*)(ws + off_Wcat);
  __bf16* PBt  = (__bf16*)(ws + off_PBt);
  float*  LD   = (float*)(ws + off_LD);
  float*  Sout = (float*)(ws + off_S);
  __bf16* Ob   = (__bf16*)(ws + off_Ob);
  float*  Y    = (float*)(ws + off_Y);
  __bf16* Ucat = (__bf16*)(ws + off_Ucat);
  __bf16* Vcat = (__bf16*)(ws + off_Vcat);

  // 1) top-k mixture weights (4 banks)
  topk_kernel<<<1, 32, 0, stream>>>(q_lg, k_lg, v_lg, g_lg, idx, wts);

  // 2) fp32 -> bf16 conversions (x, out_gate_w -> Wcat rows 4096.., out_proj_w)
  {
    size_t n = (size_t)BSTOT * DM;
    cvt_bf16_kernel<<<(unsigned)((n + 255) / 256), 256, 0, stream>>>(x, Xb, n);
    n = (size_t)HID * DM;
    cvt_bf16_kernel<<<(unsigned)((n + 255) / 256), 256, 0, stream>>>(
        og_w, Wcat + (size_t)4 * HID * DM, n);
    n = (size_t)DM * HID;
    cvt_bf16_kernel<<<(unsigned)((n + 255) / 256), 256, 0, stream>>>(op_w, PBt, n);
  }

  // 3) gather transposed bf16 copies of selected U/V primitives
  {
    size_t n = (size_t)4 * DM * 1024;
    gather_u_kernel<<<(unsigned)((n + 255) / 256), 256, 0, stream>>>(
        q_U, k_U, v_U, idx, Ucat);
    gather_v_kernel<<<(unsigned)((n + 255) / 256), 256, 0, stream>>>(
        q_V, k_V, v_V, idx, wts, Vcat);
  }

  // 4) build collapsed matrices: Wcat[bank*HID + o][d] = sum_jr Vcat*Ucat (bf16 out)
  for (int bank = 0; bank < 4; ++bank) {
    gemm_bf16_wmma<<<dim3(DM / TILE, HID / TILE), 128, 0, stream>>>(
        Vcat + (size_t)bank * HID * 1024, 1024,
        Ucat + (size_t)bank * DM * 1024, 1024,
        nullptr, Wcat + (size_t)bank * HID * DM, DM, 1024);
  }

  // 5) main GEMM: Y[8192][5120] = Xb @ Wcat^T   (q|k|v|gate|ogate)
  gemm_bf16_wmma<<<dim3(NCAT / TILE, BSTOT / TILE), 128, 0, stream>>>(
      Xb, DM, Wcat, DM, Y, nullptr, NCAT, DM);

  // 6) decay logits (N=16, too small for WMMA tiles)
  decay_kernel<<<BSTOT, 256, 0, stream>>>(x, dec_w, dec_b, LD);

  // 7) kv = sigmoid(gate)*k*v
  {
    size_t n = (size_t)BSTOT * HID;
    kv_kernel<<<(unsigned)((n + 255) / 256), 256, 0, stream>>>(Y);
  }

  // 8) gated linear recurrence
  scan_kernel<<<BB * NH, DH, 0, stream>>>(Y, LD, Sout);

  // 9) rmsnorm + out-gate, write bf16
  rms_gate_kernel<<<BSTOT, 256, 0, stream>>>(Sout, Y, rms_w, Ob);

  // 10) final projection: d_out[8192][1024] = Ob @ out_proj_w^T (PBt[d][h])
  gemm_bf16_wmma<<<dim3(DM / TILE, BSTOT / TILE), 128, 0, stream>>>(
      Ob, HID, PBt, HID, (float*)d_out, nullptr, DM, HID);
}
